// LstmCellS_54468775248255
// MI455X (gfx1250) — matrix operand order repaired and checked by reading, not compiled, not run on hardware
//
#include <hip/hip_runtime.h>
#include <hip/hip_bf16.h>

// ---- types -----------------------------------------------------------------
typedef __bf16 bf16_t;
typedef bf16_t v16bf __attribute__((ext_vector_type(16)));
typedef float  v8f   __attribute__((ext_vector_type(8)));

// ---- problem constants -----------------------------------------------------
constexpr int BN  = 256;    // batch
constexpr int S2  = 4096;   // S*S flattened state / output columns
constexpr int KI  = 512;    // input size I == hidden_size U

// LDS B slab: double-buffered, bf16, [buf][gate][p(32)][k(32)] = 2 x 8KB
#define BSLAB_T bf16_t (&Bslab)[2][4][32][32]

// ---- fast activations (emit v_exp_f32 transcendentals) ---------------------
__device__ __forceinline__ float fast_sigmoid(float v) {
    return 1.0f / (1.0f + __expf(-v));
}
__device__ __forceinline__ float fast_tanh(float v) {
    float e = __expf(2.0f * v);          // +inf/0 endpoints give +/-1 cleanly
    return 1.0f - 2.0f / (e + 1.0f);
}

// ---- cooperative B-slab stage: global f32 -> cvt -> LDS bf16 ----------------
// 256 threads stage 4 gates x 32 p-rows x 32 k (16KB f32 -> 8KB bf16).
// Thread t handles half a row: row = t>>1 (g = row>>5, p = row&31),
// k-offset = (t&1)*16 -> 64B contiguous global read, 32B LDS write.
template<int LDB>
__device__ __forceinline__ void stage_B(const float* __restrict__ W,
                                        size_t gstride, int pbase, int kb,
                                        int tid, int buf, BSLAB_T) {
    const int row  = tid >> 1;
    const int half = tid & 1;
    const int g    = row >> 5;
    const int p    = row & 31;
    const float* rp = W + (size_t)g * gstride
                        + (size_t)(pbase + p) * LDB + kb + half * 16;
    v16bf b;
#pragma unroll
    for (int e = 0; e < 16; ++e) b[e] = (bf16_t)rp[e];
    *(v16bf*)&Bslab[buf][g][p][half * 16] = b;
}

// ---- one K-segment of the fused 4-gate GEMM --------------------------------
// acc[g][ns][ps] += A x B over K in 32-wide steps.  B comes from the shared
// LDS slab (staged once per workgroup); A is converted in-register per wave.
// NEG negates A (for the -x@Wd^T term; bf16 WMMA has no A-neg modifier).
template<int K, int LDA, int LDB, bool NEG>
__device__ __forceinline__ void mma_segment(const float* __restrict__ A,
                                            const float* __restrict__ W,
                                            size_t gstride,
                                            int nbase, int pbase,
                                            int tid, int lane,
                                            v8f (&acc)[4][2][2], BSLAB_T) {
    const int l16   = lane & 15;
    const int khalf = lane >> 4;
    constexpr int NSTEP = K / 32;

    // Preload slab for step 0.
    stage_B<LDB>(W, gstride, pbase, 0, tid, 0, Bslab);
    __syncthreads();

    for (int s = 0; s < NSTEP; ++s) {
        const int kb  = s * 32;
        const int buf = s & 1;

        // Prefetch next slab into the other buffer (barrier at end of the
        // previous step guarantees all waves are done reading it).
        if (s + 1 < NSTEP)
            stage_B<LDB>(W, gstride, pbase, kb + 32, tid, buf ^ 1, Bslab);

        // A fragments: bf16 A 16x32, contiguous k order (same as B):
        // lanes 0-15: row M=lane,    K in {kb..+15}
        // lanes16-31: row M=lane-16, K in {kb+16..+31}
        v16bf af[2];
#pragma unroll
        for (int ns = 0; ns < 2; ++ns) {
            const float* rp = A + (size_t)(nbase + ns * 16 + l16) * LDA
                                + kb + khalf * 16;
            v16bf a;
#pragma unroll
            for (int e = 0; e < 8; ++e)
                a[e]     = (bf16_t)(NEG ? -rp[e]      : rp[e]);
#pragma unroll
            for (int e = 0; e < 8; ++e)
                a[8 + e] = (bf16_t)(NEG ? -rp[8 + e] : rp[8 + e]);
            af[ns] = a;
        }

        // B fragments straight from LDS, same contiguous k order as A:
        // lane holds column p = ps*16 + (lane&15); lanes<16 K=kb..kb+15,
        // lanes>=16 K=kb+16..kb+31 -> 32B contiguous ds read per lane.
#pragma unroll
        for (int g = 0; g < 4; ++g) {
            v16bf bf[2];
#pragma unroll
            for (int ps = 0; ps < 2; ++ps)
                bf[ps] = *(const v16bf*)&Bslab[buf][g][ps * 16 + l16][khalf * 16];
#pragma unroll
            for (int ns = 0; ns < 2; ++ns)
#pragma unroll
                for (int ps = 0; ps < 2; ++ps)
                    acc[g][ns][ps] = __builtin_amdgcn_wmma_f32_16x16x32_bf16(
                        false, af[ns], false, bf[ps],
                        (short)0, acc[g][ns][ps], false, false);
        }
        __syncthreads();
    }
}

// ---- fused matrix-LSTM cell kernel -----------------------------------------
// Grid: 128 blocks, each owning a 32-column slice of S2 for the full batch
// (so the 268MB Wu streams from HBM exactly once).  Block: 256 threads =
// 8 waves; wave w owns rows [32w, 32w+32) x the block's 32 columns, all gates.
__global__ __launch_bounds__(256)
void lstm_cell_s_kernel(const float* __restrict__ x,
                        const float* __restrict__ hs,
                        const float* __restrict__ hu,
                        const float* __restrict__ hc,
                        const float* __restrict__ Wd,
                        const float* __restrict__ Wu,
                        const float* __restrict__ Ww,
                        const float* __restrict__ Bd,
                        const float* __restrict__ Bu,
                        const float* __restrict__ Bw,
                        float* __restrict__ out) {
    __shared__ __align__(32) bf16_t Bslab[2][4][32][32];   // 16 KB of 320 KB

    const int tid   = threadIdx.x;
    const int lane  = tid & 31;
    const int wave  = tid >> 5;
    const int pbase = blockIdx.x * 32;
    const int nbase = wave * 32;
    const int l16   = lane & 15;
    const int khalf = lane >> 4;

    v8f acc[4][2][2];
#pragma unroll
    for (int g = 0; g < 4; ++g)
#pragma unroll
        for (int ns = 0; ns < 2; ++ns)
#pragma unroll
            for (int ps = 0; ps < 2; ++ps)
                acc[g][ns][ps] = (v8f)0.0f;

    // pre = hs@Wu^T + hu@Ww^T - x@Wd^T   (bias folded in below)
    mma_segment<S2, S2, S2, false>(hs, Wu, (size_t)S2 * S2, nbase, pbase, tid, lane, acc, Bslab);
    mma_segment<KI, KI, KI, false>(hu, Ww, (size_t)S2 * KI, nbase, pbase, tid, lane, acc, Bslab);
    mma_segment<KI, KI, KI, true >(x , Wd, (size_t)S2 * KI, nbase, pbase, tid, lane, acc, Bslab);

    // Bias sums (broadcast over batch): bias[g][p] = Bw+Bd+Bu
    float bias[4][2];
#pragma unroll
    for (int g = 0; g < 4; ++g)
#pragma unroll
        for (int ps = 0; ps < 2; ++ps) {
            size_t o = (size_t)g * S2 + pbase + ps * 16 + l16;
            bias[g][ps] = Bw[o] + Bd[o] + Bu[o];
        }

    // Epilogue: C/D 16x16 f32 layout -> element r of v8f sits at
    // row M = r + 8*(lane>=16), col N = lane&15.
#pragma unroll
    for (int ns = 0; ns < 2; ++ns)
#pragma unroll
        for (int ps = 0; ps < 2; ++ps)
#pragma unroll
            for (int r = 0; r < 8; ++r) {
                const int n = nbase + ns * 16 + khalf * 8 + r;
                const int p = pbase + ps * 16 + l16;
                const float pf = acc[0][ns][ps][r] + bias[0][ps];
                const float pi = acc[1][ns][ps][r] + bias[1][ps];
                const float po = acc[2][ns][ps][r] + bias[2][ps];
                const float pc = acc[3][ns][ps][r] + bias[3][ps];
                const float f  = fast_sigmoid(pf);
                const float i  = fast_sigmoid(pi);
                const float o  = fast_sigmoid(po);
                const float gg = fast_tanh(pc);
                const size_t idx = (size_t)n * S2 + p;
                const float c = f * hc[idx] + i * gg;
                out[idx]                   = o * fast_tanh(c);   // out_s
                out[(size_t)BN * S2 + idx] = c;                  // c
            }
}

// ---- host launch -----------------------------------------------------------
extern "C" void kernel_launch(void* const* d_in, const int* in_sizes, int n_in,
                              void* d_out, int out_size, void* d_ws, size_t ws_size,
                              hipStream_t stream) {
    const float* x  = (const float*)d_in[0];
    const float* hs = (const float*)d_in[1];
    const float* hu = (const float*)d_in[2];
    const float* hc = (const float*)d_in[3];
    const float* Wd = (const float*)d_in[4];
    const float* Wu = (const float*)d_in[5];
    const float* Ww = (const float*)d_in[6];
    const float* Bd = (const float*)d_in[7];
    const float* Bu = (const float*)d_in[8];
    const float* Bw = (const float*)d_in[9];
    float* out = (float*)d_out;

    dim3 grid(S2 / 32);   // 128 workgroups, one 32-column slice each
    dim3 block(256);      // 8 waves, each a 32x32 tile for all 4 gates
    hipLaunchKernelGGL(lstm_cell_s_kernel, grid, block, 0, stream,
                       x, hs, hu, hc, Wd, Wu, Ww, Bd, Bu, Bw, out);
}
